// DCI_loss_72043781423171
// MI455X (gfx1250) — matrix operand order repaired and checked by reading, not compiled, not run on hardware
//
#include <hip/hip_runtime.h>
#include <hip/hip_bf16.h>

typedef __attribute__((ext_vector_type(16))) _Float16 v16h;
typedef __attribute__((ext_vector_type(8)))  _Float16 v8h;
typedef __attribute__((ext_vector_type(8)))  float    v8f;
typedef __attribute__((ext_vector_type(4)))  float    fv4;

#define N_NODES   100000
#define HIDDEN    256
#define KCLUST    20
#define NC        5000
#define TPC       313            // ceil(5000/16) row tiles per cluster
#define NTILES    (KCLUST * TPC) // 6260
// ws layout: f32 S[0..5120), f32 accum at [5120], f16 Wc (5120 halves) at f32 idx 5124
#define WS_S      0
#define WS_ACC    5120
#define WS_WCH    5124           // float index; cast to _Float16*

__device__ __forceinline__ float softplusf(float x) {
    return fmaxf(x, 0.0f) + log1pf(expf(-fabsf(x)));
}

// ---------------- kernel 0: zero scratch ----------------
__global__ void k_zero(float* ws) {
    for (int i = threadIdx.x; i < 5120; i += blockDim.x) ws[WS_S + i] = 0.0f;
    if (threadIdx.x == 0) ws[WS_ACC] = 0.0f;
}

// ---------------- kernel 1: S[k,d] = sum_n h1[idx[k,n], d] ----------------
// grid (20, 25), 256 threads; each block: 200 rows, thread d accumulates dim d.
__global__ void k_cluster_sum(const float* __restrict__ h1,
                              const long long* __restrict__ ci,
                              float* __restrict__ ws) {
    const int k = blockIdx.x;
    const int d = threadIdx.x;
    const int n0 = blockIdx.y * 200;
    float acc = 0.0f;
    for (int r = 0; r < 200; ++r) {
        long long idx = ci[(size_t)k * NC + n0 + r];
        acc += h1[(size_t)idx * HIDDEN + d];   // coalesced 1KB row reads (RT: keep in L2)
    }
    atomicAdd(&ws[WS_S + k * HIDDEN + d], acc);
}

// ---------------- kernel 2: c = sigmoid(S/5000); Wc[k,d] = sum_e W[d,e]*c[k,e] via WMMA ----
// 1 block, 512 threads = 16 waves; wave w owns N-tile w (16 dims), 2 M-tiles, K=256.
// c padded to 32 rows (rows >= 20 zero) so A-fragment LDS loads are unconditional b128s.
__global__ void k_wc(const float* __restrict__ ws_in,
                     const float* __restrict__ W,
                     float* __restrict__ ws) {
    __shared__ _Float16 c_lds[32 * HIDDEN];
    for (int i = threadIdx.x; i < 32 * HIDDEN; i += blockDim.x) {
        int kk = i >> 8;
        float cv = 0.0f;
        if (kk < KCLUST) {
            float s = ws_in[WS_S + i];
            cv = 1.0f / (1.0f + expf(-(s * (1.0f / (float)NC))));
        }
        c_lds[i] = (_Float16)cv;
    }
    __syncthreads();

    const int wave = threadIdx.x >> 5;        // 0..15 -> N tile
    const int lane = threadIdx.x & 31;
    const int l16  = lane & 15;
    const int hi   = lane >> 4;               // 0 or 1

    const int dcol = wave * 16 + l16;         // B column N -> output dim d
    _Float16* wch = (_Float16*)(ws + WS_WCH);
    v8f acc0 = {}, acc1 = {};

    #pragma unroll
    for (int kt = 0; kt < 8; ++kt) {
        // B fragment: B[K=e, N=d] = W[d, e]; halves j -> e = kt*32 + hi*16 + j
        const float* wrow = W + (size_t)dcol * HIDDEN + kt * 32 + hi * 16;
        fv4 w0 = *(const fv4*)(wrow +  0);
        fv4 w1 = *(const fv4*)(wrow +  4);
        fv4 w2 = *(const fv4*)(wrow +  8);
        fv4 w3 = *(const fv4*)(wrow + 12);
        v16h bf;
        #pragma unroll
        for (int j = 0; j < 4; ++j) {
            bf[j]      = (_Float16)w0[j];
            bf[4 + j]  = (_Float16)w1[j];
            bf[8 + j]  = (_Float16)w2[j];
            bf[12 + j] = (_Float16)w3[j];
        }
        // A fragments: A[M=k, K=e] = c[k,e]; halves j -> e = kt*32 + hi*16 + j
        #pragma unroll
        for (int mt = 0; mt < 2; ++mt) {
            const int kk = mt * 16 + l16;     // matrix row M (padded rows are zero)
            const v8h* cp = (const v8h*)&c_lds[kk * HIDDEN + kt * 32 + hi * 16];
            v8h alo = cp[0];                  // K run at +0  (ds_load_b128)
            v8h ahi = cp[1];                  // K run at +8  (ds_load_b128)
            v16h af;
            #pragma unroll
            for (int j = 0; j < 8; ++j) { af[j] = alo[j]; af[8 + j] = ahi[j]; }
            if (mt == 0)
                acc0 = __builtin_amdgcn_wmma_f32_16x16x32_f16(false, af, false, bf,
                                                              (short)0, acc0, false, false);
            else
                acc1 = __builtin_amdgcn_wmma_f32_16x16x32_f16(false, af, false, bf,
                                                              (short)0, acc1, false, false);
        }
    }
    // D layout: VGPR r -> row M = r + hi*8, col N = l16; store Wc as f16 for kernel 3
    #pragma unroll
    for (int r = 0; r < 8; ++r) {
        int k0 = 0 * 16 + r + hi * 8;
        int k1 = 1 * 16 + r + hi * 8;
        if (k0 < KCLUST) wch[k0 * HIDDEN + dcol] = (_Float16)acc0[r];
        if (k1 < KCLUST) wch[k1 * HIDDEN + dcol] = (_Float16)acc1[r];
    }
}

// ---------------- kernel 3: scores + softplus, wave-per-16-row tile ----------------
// sc[k,n] = h[idx[k,n],:] . Wc[k,:] + b ; accumulate softplus(-sc1)+softplus(sc2)
__global__ void k_score(const float* __restrict__ h1,
                        const float* __restrict__ h2,
                        const long long* __restrict__ ci,
                        const float* __restrict__ ws_in,
                        const float* __restrict__ bptr,
                        float* __restrict__ ws) {
    const int wave  = threadIdx.x >> 5;
    int gtile = blockIdx.x * 8 + wave;
    const bool valid = (gtile < NTILES);
    if (!valid) gtile = 0;                    // wave-uniform clamp; EXEC stays full
    const int k  = gtile / TPC;
    const int tn = gtile % TPC;

    const int lane = threadIdx.x & 31;
    const int l16  = lane & 15;
    const int hi   = lane >> 4;

    // A rows: lanes m and m+16 share row m, split K halves
    int n = tn * 16 + l16;
    int nclamp = (n < NC) ? n : 0;
    long long idx = ci[(size_t)k * NC + nclamp];
    const float*    r1  = h1 + (size_t)idx * HIDDEN;
    const float*    r2  = h2 + (size_t)idx * HIDDEN;
    const _Float16* wc  = (const _Float16*)(ws_in + WS_WCH) + k * HIDDEN;
    const float bb = bptr[0];
    const int off = hi * 8;

    v8f acc1 = {}, acc2 = {};
    #pragma unroll
    for (int ct = 0; ct < 8; ++ct) {
        const int d0 = ct * 32 + off;
        // h1 fragment (RT: expect L2 hits from pass 1)
        fv4 p0 = *(const fv4*)(r1 + d0 + 0);
        fv4 p1 = *(const fv4*)(r1 + d0 + 4);
        fv4 q0 = *(const fv4*)(r1 + d0 + 16);
        fv4 q1 = *(const fv4*)(r1 + d0 + 20);
        // h2 fragment, single-use: non-temporal so it doesn't evict h1 from L2
        fv4 s0 = __builtin_nontemporal_load((const fv4*)(r2 + d0 + 0));
        fv4 s1 = __builtin_nontemporal_load((const fv4*)(r2 + d0 + 4));
        fv4 t0 = __builtin_nontemporal_load((const fv4*)(r2 + d0 + 16));
        fv4 t1 = __builtin_nontemporal_load((const fv4*)(r2 + d0 + 20));
        v16h a1, a2;
        #pragma unroll
        for (int j = 0; j < 4; ++j) {
            a1[j]      = (_Float16)p0[j];  a1[4 + j]  = (_Float16)p1[j];
            a1[8 + j]  = (_Float16)q0[j];  a1[12 + j] = (_Float16)q1[j];
            a2[j]      = (_Float16)s0[j];  a2[4 + j]  = (_Float16)s1[j];
            a2[8 + j]  = (_Float16)t0[j];  a2[12 + j] = (_Float16)t1[j];
        }
        // B: broadcast Wc[k] down every column; halves j<8 -> K = ct*32 + hi*8 + j, j>=8 -> K = ct*32 + 16 + hi*8 + (j-8)
        const v8h* wcc = (const v8h*)(wc + ct * 32 + hi * 8);
        v8h b0 = wcc[0];                      // global_load_b128, f16 direct, no cvts
        v8h b1 = wcc[2];
        v16h bf;
        #pragma unroll
        for (int j = 0; j < 8; ++j) { bf[j] = b0[j]; bf[8 + j] = b1[j]; }
        acc1 = __builtin_amdgcn_wmma_f32_16x16x32_f16(false, a1, false, bf,
                                                      (short)0, acc1, false, false);
        acc2 = __builtin_amdgcn_wmma_f32_16x16x32_f16(false, a2, false, bf,
                                                      (short)0, acc2, false, false);
    }

    // every D column identical; lanes 0 and 16 cover rows 0-7 / 8-15
    float s = 0.0f;
    if (valid && l16 == 0) {
        #pragma unroll
        for (int r = 0; r < 8; ++r) {
            int row = r + hi * 8;
            int n2 = tn * 16 + row;
            if (n2 < NC) {
                s += softplusf(-(acc1[r] + bb));   // positives, label 1
                s += softplusf(  acc2[r] + bb );   // negatives, label 0
            }
        }
    }
    // block reduction: lane16 -> lane0, wave partials -> LDS, one atomic per block
    __shared__ float red[8];
    s += __shfl_down(s, 16, 32);
    if (lane == 0) red[wave] = s;
    __syncthreads();
    if (threadIdx.x == 0) {
        float t = 0.0f;
        #pragma unroll
        for (int i = 0; i < 8; ++i) t += red[i];
        atomicAdd(&ws[WS_ACC], t);
    }
}

// ---------------- kernel 4: finalize ----------------
__global__ void k_final(const float* __restrict__ ws, float* __restrict__ out) {
    out[0] = ws[WS_ACC] * (0.5f / ((float)KCLUST * (float)NC));
}

extern "C" void kernel_launch(void* const* d_in, const int* in_sizes, int n_in,
                              void* d_out, int out_size, void* d_ws, size_t ws_size,
                              hipStream_t stream) {
    (void)in_sizes; (void)n_in; (void)out_size; (void)ws_size;
    const float*     h1 = (const float*)d_in[0];
    const float*     h2 = (const float*)d_in[1];
    const long long* ci = (const long long*)d_in[2];
    const float*     W  = (const float*)d_in[3];
    const float*     b  = (const float*)d_in[4];
    float* ws  = (float*)d_ws;
    float* out = (float*)d_out;

    k_zero<<<1, 256, 0, stream>>>(ws);
    k_cluster_sum<<<dim3(KCLUST, 25), 256, 0, stream>>>(h1, ci, ws);
    k_wc<<<1, 512, 0, stream>>>(ws, W, ws);
    k_score<<<(NTILES + 7) / 8, 256, 0, stream>>>(h1, h2, ci, ws, b, ws);
    k_final<<<1, 1, 0, stream>>>(ws, out);
}
